// AlignmentQFormer_54949811585154
// MI455X (gfx1250) — matrix operand order repaired and checked by reading, not compiled, not run on hardware
//
#include <hip/hip_runtime.h>
#include <math.h>

// ---------------- problem constants ----------------
#define CB   8
#define CT   1500
#define CN   128
#define CD   256
#define CH   8
#define DH   32          // CD / CH
#define QL   4
#define QT   5           // QL + 1
#define CNQ  (CN * QT)   // 640
#define FFND (4 * CD)    // 1024
#define OUTD 64
#define CTXW 5
#define NEGV (-1e9f)
#define EPSV 1e-5f
#define SCL_DH 0.17677669529663689f   // 1/sqrt(32)

typedef __attribute__((ext_vector_type(16))) _Float16 v16h;
typedef __attribute__((ext_vector_type(8)))  float    v8f;

__device__ __forceinline__ v8f wmma_f16(v16h a, v16h b, v8f c) {
  // D = A(16x32 f16) * B(32x16 f16) + C(16x16 f32)
  return __builtin_amdgcn_wmma_f32_16x16x32_f16(false, a, false, b, (short)0, c, false, false);
}

// gfx1250 prefetch (lowers to global_prefetch_b8)
__device__ __forceinline__ void pf(const void* p) { __builtin_prefetch(p, 0, 1); }

enum { ACT_NONE = 0, ACT_GELU = 1 };

// ============================================================
// Generic WMMA GEMM: C = act( A(MxK) * B^T + bias ) [+resid]
//   BTR=true : Bm is [Nout][K] row-major (weight style, x @ W.T)
//   BTR=false: Bm is [K][Nout] row-major (plain second operand)
//   SQB: square B elements on load (for E[x^2] pooling)
//   RES: add resid (same layout as C) after activation
//   SAN: nan_to_num on the matmul+bias result (before resid)
// One wave per 16(M) x 64(N) strip: A-fragment loaded once per
// K-step and reused across 4 B-fragments / 4 WMMAs. K tail handled.
// Requires Nout % 64 == 0 (true for all uses: 256, 1024, 64).
// ============================================================
template<int ACT, bool BTR, bool SQB, bool RES, bool SAN>
__global__ __launch_bounds__(32)
void gemm_wmma(const float* __restrict__ A, int lda, long strideA,
               const float* __restrict__ Bm, int ldb, long strideB,
               const float* __restrict__ bias,
               const float* __restrict__ Rz, long strideR,
               float* __restrict__ C, int ldc, long strideC,
               int M, int K, int Nout)
{
  int bm = blockIdx.x, bn = blockIdx.y, bz = blockIdx.z;
  A  += (long)bz * strideA;
  Bm += (long)bz * strideB;
  C  += (long)bz * strideC;
  const float* R = RES ? (Rz + (long)bz * strideR) : (const float*)0;

  int lane  = threadIdx.x;
  int l16   = lane & 15;
  int hi    = lane >> 4;
  int koffA = hi * 16;   // A frag: K = {koffA..+15}
  int koffB = hi * 16;   // B frag: K = {koffB..+15}

  int mrow = bm * 16 + l16; if (mrow >= M) mrow = M - 1;
  const float* arow = A + (long)mrow * lda;

  int ncol[4];
  const float* brow[4];
#pragma unroll
  for (int nt = 0; nt < 4; ++nt) {
    int nc = bn * 64 + nt * 16 + l16;
    if (nc >= Nout) nc = Nout - 1;
    ncol[nt] = nc;
    brow[nt] = BTR ? (Bm + (long)nc * ldb) : (const float*)0;
  }

  v8f acc[4] = {};
  for (int k0 = 0; k0 < K; k0 += 32) {
    // prefetch next K tile (L2-resident working set; hide VMEM latency)
    if (k0 + 32 < K) {
      pf(arow + k0 + 32);
      if (BTR) pf(brow[0] + k0 + 32);
      else     pf(Bm + (long)(k0 + 32) * ldb + ncol[0]);
    }
    v16h af;
#pragma unroll
    for (int i = 0; i < 8; ++i) {
      int ka1 = k0 + koffA + i;
      int ka2 = k0 + koffA + 8 + i;
      af[i]     = (_Float16)((ka1 < K) ? arow[ka1] : 0.f);
      af[8 + i] = (_Float16)((ka2 < K) ? arow[ka2] : 0.f);
    }
#pragma unroll
    for (int nt = 0; nt < 4; ++nt) {
      v16h bf;
#pragma unroll
      for (int i = 0; i < 16; ++i) {
        int kb = k0 + koffB + i;
        float bv = 0.f;
        if (kb < K) bv = BTR ? brow[nt][kb] : Bm[(long)kb * ldb + ncol[nt]];
        if (SQB) bv *= bv;
        bf[i] = (_Float16)bv;
      }
      acc[nt] = wmma_f16(af, bf, acc[nt]);
    }
  }

  int mbase = bm * 16 + hi * 8;
#pragma unroll
  for (int nt = 0; nt < 4; ++nt) {
    int no = bn * 64 + nt * 16 + l16;
    if (no >= Nout) continue;
#pragma unroll
    for (int j = 0; j < 8; ++j) {
      int mo = mbase + j;
      if (mo < M) {
        float v = acc[nt][j];
        if (bias) v += bias[no];
        if (SAN) { if (!(v == v)) v = 0.f; }
        if (ACT == ACT_GELU) v = 0.5f * v * (1.f + erff(v * 0.70710678118654752f));
        if (RES) v += R[(long)mo * ldc + no];
        C[(long)mo * ldc + no] = v;
      }
    }
  }
}

// ============================================================
// Alignment preprocessing: per-batch block.
// idx[t] = argmax_n alignment; dur[n]; rel_pos[t]
// ============================================================
__global__ __launch_bounds__(256)
void align_kernel(const float* __restrict__ alig, int* __restrict__ idxb,
                  float* __restrict__ durb, float* __restrict__ relout)
{
  int b = blockIdx.x, tid = threadIdx.x;
  __shared__ int   sidx[CT];
  __shared__ float sdur[CN];
  for (int t = tid; t < CT; t += 256) {
    const float* arow = alig + ((long)b * CT + t) * CN;
    int best = 0; float bv = arow[0];
    for (int n2 = 1; n2 < CN; ++n2) { float v = arow[n2]; if (v > bv) { bv = v; best = n2; } }
    sidx[t] = best; idxb[b * CT + t] = best;
  }
  __syncthreads();
  for (int n2 = tid; n2 < CN; n2 += 256) {
    int c = 0;
    for (int t = 0; t < CT; ++t) c += (sidx[t] == n2);
    sdur[n2] = (float)c; durb[b * CN + n2] = (float)c;
  }
  __syncthreads();
  for (int t = tid; t < CT; t += 256) {
    int me = sidx[t], c = 0;
    for (int tt = 0; tt <= t; ++tt) c += (sidx[tt] == me);
    float dn = sdur[me];
    float r  = ((float)c - 1.f) / fmaxf(dn - 1.f, 1.f);
    relout[b * CT + t] = fminf(fmaxf(r, 0.f), 1.f);
  }
}

// additive cross-attn mask (B,N,T): 0 if window hit (or empty row), else NEG
__global__ __launch_bounds__(256)
void mask_kernel(const int* __restrict__ idxb, const float* __restrict__ durb,
                 float* __restrict__ addm)
{
  int bn = blockIdx.x; int b = bn / CN, n = bn % CN;
  float durv = durb[bn];
  for (int t = threadIdx.x; t < CT; t += 256) {
    int lo = t - CTXW < 0 ? 0 : t - CTXW;
    int hh = t + CTXW >= CT ? CT - 1 : t + CTXW;
    bool hit = false;
    for (int tt = lo; tt <= hh; ++tt) if (idxb[b * CT + tt] == n) { hit = true; break; }
    addm[(long)bn * CT + t] = (durv == 0.f) ? 0.f : (hit ? 0.f : NEGV);
  }
}

// sinusoidal features (B*T, D)
__global__ __launch_bounds__(256)
void feats_kernel(const float* __restrict__ relout, float* __restrict__ feats)
{
  int row = blockIdx.x, d = threadIdx.x;
  float rel  = relout[row];
  int  halfd = CD / 2;
  int  k     = (d < halfd) ? d : d - halfd;
  float fr   = __expf(-9.210340371976184f * (float)k / (float)(halfd - 1));
  float a    = rel * fr;
  feats[(long)row * CD + d] = (d < halfd) ? sinf(a) : cosf(a);
}

// queries = concat([phon, protos]) per phoneme
__global__ __launch_bounds__(256)
void queries_kernel(const float* __restrict__ phon, const float* __restrict__ proto,
                    float* __restrict__ queries)
{
  int row = blockIdx.x, d = threadIdx.x;
  int b = row / CNQ, q = row % CNQ;
  int n = q / QT, j = q % QT;
  float v = (j == 0) ? phon[((long)b * CN + n) * CD + d] : proto[(long)(j - 1) * CD + d];
  queries[(long)row * CD + d] = v;
}

// LayerNorm over D=256, one block per row
__global__ __launch_bounds__(256)
void ln_kernel(const float* __restrict__ X, const float* __restrict__ g,
               const float* __restrict__ bb, float* __restrict__ Y)
{
  __shared__ float red[256];
  int row = blockIdx.x, d = threadIdx.x;
  float x = X[(long)row * CD + d];
  red[d] = x; __syncthreads();
  for (int s = 128; s > 0; s >>= 1) { if (d < s) red[d] += red[d + s]; __syncthreads(); }
  float mean = red[0] / (float)CD; __syncthreads();
  float dx = x - mean;
  red[d] = dx * dx; __syncthreads();
  for (int s = 128; s > 0; s >>= 1) { if (d < s) red[d] += red[d + s]; __syncthreads(); }
  float var = red[0] / (float)CD;
  Y[(long)row * CD + d] = dx * rsqrtf(var + EPSV) * g[d] + bb[d];
}

// ============================================================
// Flash cross-attention (layer 0): one wave per (b, h, 16-query tile).
// Streams T keys in chunks of 32: 2 score WMMAs + online softmax via
// LDS + 2 PV WMMAs per chunk. Output: head-concat context (B,NQ,D).
// ============================================================
__global__ __launch_bounds__(32)
void flash_kernel(const float* __restrict__ Qc, const float* __restrict__ Kc,
                  const float* __restrict__ Vc, const float* __restrict__ addm,
                  float* __restrict__ ctx)
{
  int qt = blockIdx.x, h = blockIdx.y, b = blockIdx.z;
  int q0 = qt * 16;
  int lane = threadIdx.x;
  int l16  = lane & 15;
  int hi   = lane >> 4;
  int koffA = hi * 16, koffB = hi * 16;

  __shared__ float    s_s[16][32];
  __shared__ _Float16 s_p[16][32];
  __shared__ float    s_fac[16];
  __shared__ float    s_linv[16];

  v16h aq;
  {
    const float* qp = Qc + ((long)b * CNQ + q0 + l16) * CD + h * DH;
#pragma unroll
    for (int i = 0; i < 8; ++i) {
      aq[i]     = (_Float16)qp[koffA + i];
      aq[8 + i] = (_Float16)qp[koffA + 8 + i];
    }
  }

  v8f acc0 = {}, acc1 = {};
  float m_r = -1e30f, l_r = 0.f;

  for (int key0 = 0; key0 < CT; key0 += 32) {
    // ---- score tiles: S = Q(16x32) x K^T(32x16) twice ----
    v16h bk0, bk1;
    {
      int t0 = key0 + l16;      if (t0 >= CT) t0 = CT - 1;
      int t1 = key0 + 16 + l16; if (t1 >= CT) t1 = CT - 1;
      const float* kp0 = Kc + ((long)b * CT + t0) * CD + h * DH;
      const float* kp1 = Kc + ((long)b * CT + t1) * CD + h * DH;
      if (key0 + 32 < CT) {               // prefetch next chunk's K/V rows
        pf(kp0 + 32 * CD);
        pf(Vc + ((long)b * CT + t0 + 32) * CD + h * DH);
      }
#pragma unroll
      for (int i = 0; i < 16; ++i) {
        bk0[i] = (_Float16)kp0[koffB + i];
        bk1[i] = (_Float16)kp1[koffB + i];
      }
    }
    v8f s0 = {}, s1 = {};
    s0 = wmma_f16(aq, bk0, s0);
    s1 = wmma_f16(aq, bk1, s1);
#pragma unroll
    for (int j = 0; j < 8; ++j) {
      int r = hi * 8 + j;
      s_s[r][l16]      = s0[j];
      s_s[r][16 + l16] = s1[j];
    }
    __syncthreads();

    // ---- per-row online softmax (lanes 0..15, one row each) ----
    if (lane < 16) {
      int r = lane, q = q0 + r, ng = q / QT;
      const float* mrow = addm + ((long)b * CN + ng) * CT;
      float vals[32];
      float cm = -1e30f;
#pragma unroll
      for (int kk = 0; kk < 32; ++kk) {
        int t = key0 + kk;
        float v = (t < CT) ? (s_s[r][kk] * SCL_DH + mrow[t]) : -1e30f;
        vals[kk] = v; cm = fmaxf(cm, v);
      }
      float nm  = fmaxf(m_r, cm);
      float fac = __expf(m_r - nm);
      float sum = 0.f;
#pragma unroll
      for (int kk = 0; kk < 32; ++kk) {
        float pv = __expf(vals[kk] - nm);
        sum += pv;
        s_p[r][kk] = (_Float16)pv;
      }
      l_r = l_r * fac + sum;
      m_r = nm;
      s_fac[r] = fac;
    }
    __syncthreads();

    // ---- rescale accumulators, P x V ----
#pragma unroll
    for (int j = 0; j < 8; ++j) {
      float f = s_fac[hi * 8 + j];
      acc0[j] *= f; acc1[j] *= f;
    }
    v16h ap;
#pragma unroll
    for (int i = 0; i < 8; ++i) {
      ap[i]     = s_p[l16][koffA + i];
      ap[8 + i] = s_p[l16][koffA + 8 + i];
    }
    v16h bv0, bv1;
#pragma unroll
    for (int i = 0; i < 16; ++i) {
      int t = key0 + koffB + i; if (t >= CT) t = CT - 1;
      const float* vp = Vc + ((long)b * CT + t) * CD + h * DH;
      bv0[i] = (_Float16)vp[l16];
      bv1[i] = (_Float16)vp[16 + l16];
    }
    acc0 = wmma_f16(ap, bv0, acc0);
    acc1 = wmma_f16(ap, bv1, acc1);
    __syncthreads();
  }

  if (lane < 16) s_linv[lane] = (l_r > 0.f) ? 1.f / l_r : 0.f;
  __syncthreads();
#pragma unroll
  for (int j = 0; j < 8; ++j) {
    int r = hi * 8 + j;
    float inv = s_linv[r];
    float* op = ctx + ((long)b * CNQ + q0 + r) * CD + h * DH;
    op[l16]      = acc0[j] * inv;
    op[16 + l16] = acc1[j] * inv;
  }
}

// ============================================================
// Block-diagonal self-attention: one block per (b, phoneme group).
// Group mask => plain 5x5 softmax within the group.
// ============================================================
__global__ __launch_bounds__(256)
void selfattn_kernel(const float* __restrict__ Qs, const float* __restrict__ Ks,
                     const float* __restrict__ Vs, float* __restrict__ ctx)
{
  int bn = blockIdx.x; int b = bn / CN, n = bn % CN;
  int g0 = n * QT;
  int tid = threadIdx.x;
  __shared__ float qsm[QT][CD], ksm[QT][CD], vsm[QT][CD];
  __shared__ float sc[CH][QT][QT];

  for (int e = tid; e < QT * CD; e += 256) {
    int i = e / CD, d = e % CD;
    long base = ((long)b * CNQ + g0 + i) * CD + d;
    qsm[i][d] = Qs[base]; ksm[i][d] = Ks[base]; vsm[i][d] = Vs[base];
  }
  __syncthreads();
  if (tid < CH * QT * QT) {
    int h = tid / (QT * QT), r = tid % (QT * QT);
    int i = r / QT, j = r % QT;
    float dot = 0.f;
    for (int d2 = 0; d2 < DH; ++d2) dot += qsm[i][h * DH + d2] * ksm[j][h * DH + d2];
    sc[h][i][j] = dot * SCL_DH;
  }
  __syncthreads();
  if (tid < CH * QT) {
    int h = tid / QT, i = tid % QT;
    float m = -1e30f;
    for (int j = 0; j < QT; ++j) m = fmaxf(m, sc[h][i][j]);
    float s = 0.f;
    for (int j = 0; j < QT; ++j) { float e = __expf(sc[h][i][j] - m); sc[h][i][j] = e; s += e; }
    float inv = 1.f / s;
    for (int j = 0; j < QT; ++j) sc[h][i][j] *= inv;
  }
  __syncthreads();
  int h = tid >> 5, d = tid & 31;
  for (int i = 0; i < QT; ++i) {
    float o = 0.f;
    for (int j = 0; j < QT; ++j) o += sc[h][i][j] * vsm[j][h * DH + d];
    ctx[((long)b * CNQ + g0 + i) * CD + h * DH + d] = o;
  }
}

// gather first query of each group
__global__ __launch_bounds__(256)
void gather_first_kernel(const float* __restrict__ queries, float* __restrict__ qfirst)
{
  int row = blockIdx.x, d = threadIdx.x;
  int b = row / CN, n = row % CN;
  qfirst[(long)row * CD + d] = queries[((long)b * CNQ + n * QT) * CD + d];
}

// ============================================================
// Layer-1 path: per (b,n) compute per-head softmax over T keys,
// head-average, add mask again, second softmax -> weights + sum.
// ============================================================
__global__ __launch_bounds__(256)
void attn2_kernel(const float* __restrict__ Qf, const float* __restrict__ Kc,
                  const float* __restrict__ addm, float* __restrict__ wout,
                  float* __restrict__ swout)
{
  int bn = blockIdx.x; int b = bn / CN;
  int tid = threadIdx.x;
  __shared__ float qv[CD];
  __shared__ float sc[CH][CT];
  __shared__ float lg[CT];
  __shared__ float red[256];
  __shared__ float hm[CH], hl[CH];

  qv[tid] = Qf[(long)bn * CD + tid];
  __syncthreads();
  const float* mrow = addm + (long)bn * CT;

  for (int t = tid; t < CT; t += 256) {
    const float* krow = Kc + ((long)b * CT + t) * CD;
    float mv = mrow[t];
    for (int h = 0; h < CH; ++h) {
      float dot = 0.f;
      for (int d2 = 0; d2 < DH; ++d2) dot += qv[h * DH + d2] * krow[h * DH + d2];
      sc[h][t] = dot * SCL_DH + mv;
    }
  }
  __syncthreads();

  for (int h = 0; h < CH; ++h) {
    float lm = -1e30f;
    for (int t = tid; t < CT; t += 256) lm = fmaxf(lm, sc[h][t]);
    red[tid] = lm; __syncthreads();
    for (int s = 128; s > 0; s >>= 1) { if (tid < s) red[tid] = fmaxf(red[tid], red[tid + s]); __syncthreads(); }
    if (tid == 0) hm[h] = red[0];
    __syncthreads();
    float ls = 0.f;
    for (int t = tid; t < CT; t += 256) ls += __expf(sc[h][t] - hm[h]);
    red[tid] = ls; __syncthreads();
    for (int s = 128; s > 0; s >>= 1) { if (tid < s) red[tid] += red[tid + s]; __syncthreads(); }
    if (tid == 0) hl[h] = red[0];
    __syncthreads();
  }

  for (int t = tid; t < CT; t += 256) {
    float a = 0.f;
    for (int h = 0; h < CH; ++h) a += __expf(sc[h][t] - hm[h]) / hl[h];
    lg[t] = a * (1.f / (float)CH) + mrow[t];
  }
  __syncthreads();

  float lm = -1e30f;
  for (int t = tid; t < CT; t += 256) lm = fmaxf(lm, lg[t]);
  red[tid] = lm; __syncthreads();
  for (int s = 128; s > 0; s >>= 1) { if (tid < s) red[tid] = fmaxf(red[tid], red[tid + s]); __syncthreads(); }
  float gm = red[0]; __syncthreads();
  float ls = 0.f;
  for (int t = tid; t < CT; t += 256) ls += __expf(lg[t] - gm);
  red[tid] = ls; __syncthreads();
  for (int s = 128; s > 0; s >>= 1) { if (tid < s) red[tid] += red[tid + s]; __syncthreads(); }
  float gl = red[0]; __syncthreads();
  float inv = (gl > 0.f) ? 1.f / gl : 0.f;

  float swl = 0.f;
  for (int t = tid; t < CT; t += 256) {
    float w = __expf(lg[t] - gm) * inv;
    if (!(w == w)) w = 0.f;
    wout[(long)bn * CT + t] = w;
    swl += w;
  }
  red[tid] = swl; __syncthreads();
  for (int s = 128; s > 0; s >>= 1) { if (tid < s) red[tid] += red[tid + s]; __syncthreads(); }
  if (tid == 0) swout[bn] = red[0];
}

// var/z: z = [mu, sqrt(max(ex2 - 2mu^2 + mu^2*sw, 0) + 1e-5)]
__global__ __launch_bounds__(256)
void varz_kernel(const float* __restrict__ mu, const float* __restrict__ ex2,
                 const float* __restrict__ sw, float* __restrict__ zb)
{
  int row = blockIdx.x, d = threadIdx.x;
  float m  = mu[(long)row * CD + d];
  float e2 = ex2[(long)row * CD + d];
  float s  = sw[row];
  float var = fmaxf(e2 - 2.f * m * m + m * m * s, 0.f);
  zb[(long)row * (2 * CD) + d]      = m;
  zb[(long)row * (2 * CD) + CD + d] = sqrtf(var + 1e-5f);
}

// ============================================================
// Host launcher
// ============================================================
extern "C" void kernel_launch(void* const* d_in, const int* in_sizes, int n_in,
                              void* d_out, int out_size, void* d_ws, size_t ws_size,
                              hipStream_t stream)
{
  (void)in_sizes; (void)n_in; (void)out_size; (void)ws_size;

  const float* mel   = (const float*)d_in[0];
  const float* alig  = (const float*)d_in[1];
  const float* phon  = (const float*)d_in[2];

  // params leaves in JAX pytree order (dict keys sorted alphabetically):
  // layers[i]: ca{bk,bo,bq,bv,wk,wo,wq,wv}, ffn_b1, ffn_b2, ffn_w1, ffn_w2,
  //            ln_ca_b, ln_ca_g, ln_ff_b, ln_ff_g, ln_sa_b, ln_sa_g,
  //            sa{bk,bo,bq,bv,wk,wo,wq,wv}
  // then pos_b, pos_w, proj_b, proj_w, query_proto
  struct MhaP { const float *bk,*bo,*bq,*bv,*wk,*wo,*wq,*wv; };
  struct LayerP { MhaP ca; const float *fb1,*fb2,*fw1,*fw2,
                  *lncab,*lncag,*lnffb,*lnffg,*lnsab,*lnsag; MhaP sa; };
  LayerP lp[2];
  int p = 3;
  for (int i = 0; i < 2; ++i) {
    lp[i].ca = { (const float*)d_in[p+0],(const float*)d_in[p+1],(const float*)d_in[p+2],
                 (const float*)d_in[p+3],(const float*)d_in[p+4],(const float*)d_in[p+5],
                 (const float*)d_in[p+6],(const float*)d_in[p+7] }; p += 8;
    lp[i].fb1=(const float*)d_in[p++]; lp[i].fb2=(const float*)d_in[p++];
    lp[i].fw1=(const float*)d_in[p++]; lp[i].fw2=(const float*)d_in[p++];
    lp[i].lncab=(const float*)d_in[p++]; lp[i].lncag=(const float*)d_in[p++];
    lp[i].lnffb=(const float*)d_in[p++]; lp[i].lnffg=(const float*)d_in[p++];
    lp[i].lnsab=(const float*)d_in[p++]; lp[i].lnsag=(const float*)d_in[p++];
    lp[i].sa = { (const float*)d_in[p+0],(const float*)d_in[p+1],(const float*)d_in[p+2],
                 (const float*)d_in[p+3],(const float*)d_in[p+4],(const float*)d_in[p+5],
                 (const float*)d_in[p+6],(const float*)d_in[p+7] }; p += 8;
  }
  const float* pos_b  = (const float*)d_in[p++];
  const float* pos_w  = (const float*)d_in[p++];
  const float* proj_b = (const float*)d_in[p++];
  const float* proj_w = (const float*)d_in[p++];
  const float* qproto = (const float*)d_in[p++];

  // ---- outputs: pooled (B,N,64) | rel_pos (B,T) | weights (B,N,T) ----
  float* outF    = (float*)d_out;
  float* poolOut = outF;
  float* relOut  = outF + (long)CB * CN * OUTD;
  float* wOut    = relOut + (long)CB * CT;

  // ---- workspace arena (~84 MB of floats) ----
  float* W = (float*)d_ws;
  size_t off = 0;
  auto alloc = [&](size_t n) { float* q = W + off; off += n; return q; };
  float* addm    = alloc((size_t)CB * CN * CT);       // additive mask
  int*   idxb    = (int*)alloc((size_t)CB * CT);
  float* durb    = alloc((size_t)CB * CN);
  float* feats   = alloc((size_t)CB * CT * CD);       // reused: self-attn K/V
  float* melpos  = alloc((size_t)CB * CT * CD);
  float* queries = alloc((size_t)CB * CNQ * CD);
  float* qn      = alloc((size_t)CB * CNQ * CD);
  float* Qc      = alloc((size_t)CB * CNQ * CD);      // reused: self-attn Q
  float* KV      = alloc(2 * (size_t)CB * CT * CD);   // Kc|Vc ; reused: FFN hidden, layer-1 K
  float* ctx     = alloc((size_t)CB * CNQ * CD);
  float* qfirst  = alloc((size_t)CB * CN * CD);
  float* qfn     = alloc((size_t)CB * CN * CD);
  float* Qf      = alloc((size_t)CB * CN * CD);
  float* mu      = alloc((size_t)CB * CN * CD);
  float* ex2     = alloc((size_t)CB * CN * CD);
  float* zb      = alloc((size_t)CB * CN * 2 * CD);
  float* sw      = alloc((size_t)CB * CN);
  float* Kc = KV;
  float* Vc = KV + (size_t)CB * CT * CD;
  float* hidden = KV;                                  // overlays dead Kc|Vc
  float* Ks = feats;                                   // overlays dead feats
  float* Vs = feats + (size_t)CB * CNQ * CD;

  // ---- preprocessing ----
  align_kernel<<<CB, 256, 0, stream>>>(alig, idxb, durb, relOut);
  mask_kernel<<<CB * CN, 256, 0, stream>>>(idxb, durb, addm);
  feats_kernel<<<CB * CT, 256, 0, stream>>>(relOut, feats);

  // mel_pos = feats @ pos_w^T + pos_b + mel
  gemm_wmma<ACT_NONE, true, false, true, false><<<dim3((CB*CT)/16, CD/64, 1), 32, 0, stream>>>(
      feats, CD, 0, pos_w, CD, 0, pos_b, mel, 0, melpos, CD, 0, CB*CT, CD, CD);

  queries_kernel<<<CB * CNQ, 256, 0, stream>>>(phon, qproto, queries);

  // ================= layer 0 (full) =================
  ln_kernel<<<CB * CNQ, 256, 0, stream>>>(queries, lp[0].lncag, lp[0].lncab, qn);
  gemm_wmma<ACT_NONE, true, false, false, false><<<dim3((CB*CNQ)/16, CD/64, 1), 32, 0, stream>>>(
      qn, CD, 0, lp[0].ca.wq, CD, 0, lp[0].ca.bq, nullptr, 0, Qc, CD, 0, CB*CNQ, CD, CD);
  gemm_wmma<ACT_NONE, true, false, false, false><<<dim3((CB*CT)/16, CD/64, 1), 32, 0, stream>>>(
      melpos, CD, 0, lp[0].ca.wk, CD, 0, lp[0].ca.bk, nullptr, 0, Kc, CD, 0, CB*CT, CD, CD);
  gemm_wmma<ACT_NONE, true, false, false, false><<<dim3((CB*CT)/16, CD/64, 1), 32, 0, stream>>>(
      melpos, CD, 0, lp[0].ca.wv, CD, 0, lp[0].ca.bv, nullptr, 0, Vc, CD, 0, CB*CT, CD, CD);

  flash_kernel<<<dim3(CNQ/16, CH, CB), 32, 0, stream>>>(Qc, Kc, Vc, addm, ctx);

  // queries += ctx @ wo^T + bo
  gemm_wmma<ACT_NONE, true, false, true, false><<<dim3((CB*CNQ)/16, CD/64, 1), 32, 0, stream>>>(
      ctx, CD, 0, lp[0].ca.wo, CD, 0, lp[0].ca.bo, queries, 0, queries, CD, 0, CB*CNQ, CD, CD);

  // self-attention
  ln_kernel<<<CB * CNQ, 256, 0, stream>>>(queries, lp[0].lnsag, lp[0].lnsab, qn);
  gemm_wmma<ACT_NONE, true, false, false, false><<<dim3((CB*CNQ)/16, CD/64, 1), 32, 0, stream>>>(
      qn, CD, 0, lp[0].sa.wq, CD, 0, lp[0].sa.bq, nullptr, 0, Qc, CD, 0, CB*CNQ, CD, CD);
  gemm_wmma<ACT_NONE, true, false, false, false><<<dim3((CB*CNQ)/16, CD/64, 1), 32, 0, stream>>>(
      qn, CD, 0, lp[0].sa.wk, CD, 0, lp[0].sa.bk, nullptr, 0, Ks, CD, 0, CB*CNQ, CD, CD);
  gemm_wmma<ACT_NONE, true, false, false, false><<<dim3((CB*CNQ)/16, CD/64, 1), 32, 0, stream>>>(
      qn, CD, 0, lp[0].sa.wv, CD, 0, lp[0].sa.bv, nullptr, 0, Vs, CD, 0, CB*CNQ, CD, CD);
  selfattn_kernel<<<CB * CN, 256, 0, stream>>>(Qc, Ks, Vs, ctx);
  // queries += nan_to_num(ctx @ wo^T + bo)
  gemm_wmma<ACT_NONE, true, false, true, true><<<dim3((CB*CNQ)/16, CD/64, 1), 32, 0, stream>>>(
      ctx, CD, 0, lp[0].sa.wo, CD, 0, lp[0].sa.bo, queries, 0, queries, CD, 0, CB*CNQ, CD, CD);

  // FFN
  ln_kernel<<<CB * CNQ, 256, 0, stream>>>(queries, lp[0].lnffg, lp[0].lnffb, qn);
  gemm_wmma<ACT_GELU, true, false, false, false><<<dim3((CB*CNQ)/16, FFND/64, 1), 32, 0, stream>>>(
      qn, CD, 0, lp[0].fw1, CD, 0, lp[0].fb1, nullptr, 0, hidden, FFND, 0, CB*CNQ, CD, FFND);
  gemm_wmma<ACT_NONE, true, false, true, false><<<dim3((CB*CNQ)/16, CD/64, 1), 32, 0, stream>>>(
      hidden, FFND, 0, lp[0].fw2, FFND, 0, lp[0].fb2, queries, 0, queries, CD, 0, CB*CNQ, FFND, CD);

  // ================= layer 1 (probs-only path) =================
  gather_first_kernel<<<CB * CN, 256, 0, stream>>>(queries, qfirst);
  ln_kernel<<<CB * CN, 256, 0, stream>>>(qfirst, lp[1].lncag, lp[1].lncab, qfn);
  gemm_wmma<ACT_NONE, true, false, false, false><<<dim3((CB*CN)/16, CD/64, 1), 32, 0, stream>>>(
      qfn, CD, 0, lp[1].ca.wq, CD, 0, lp[1].ca.bq, nullptr, 0, Qf, CD, 0, CB*CN, CD, CD);
  gemm_wmma<ACT_NONE, true, false, false, false><<<dim3((CB*CT)/16, CD/64, 1), 32, 0, stream>>>(
      melpos, CD, 0, lp[1].ca.wk, CD, 0, lp[1].ca.bk, nullptr, 0, Kc, CD, 0, CB*CT, CD, CD);

  attn2_kernel<<<CB * CN, 256, 0, stream>>>(Qf, Kc, addm, wOut, sw);

  // ---- pooled stats: mu = w @ mel, ex2 = w @ mel^2 (batched over B) ----
  gemm_wmma<ACT_NONE, false, false, false, false><<<dim3(CN/16, CD/64, CB), 32, 0, stream>>>(
      wOut, CT, (long)CN*CT, mel, CD, (long)CT*CD, nullptr, nullptr, 0,
      mu, CD, (long)CN*CD, CN, CT, CD);
  gemm_wmma<ACT_NONE, false, true, false, false><<<dim3(CN/16, CD/64, CB), 32, 0, stream>>>(
      wOut, CT, (long)CN*CT, mel, CD, (long)CT*CD, nullptr, nullptr, 0,
      ex2, CD, (long)CN*CD, CN, CT, CD);

  varz_kernel<<<CB * CN, 256, 0, stream>>>(mu, ex2, sw, zb);

  // pooled = z @ proj_w^T + proj_b
  gemm_wmma<ACT_NONE, true, false, false, false><<<dim3((CB*CN)/16, OUTD/64, 1), 32, 0, stream>>>(
      zb, 2*CD, 0, proj_w, 2*CD, 0, proj_b, nullptr, 0, poolOut, OUTD, 0, CB*CN, 2*CD, OUTD);
}
